// DepFormerSlice_1941325218434
// MI455X (gfx1250) — compile-verified, matrix operand order checked by reading, not run on hardware
//
#include <hip/hip_runtime.h>
#include <hip/hip_bf16.h>
#include <math.h>

// ---------------- model constants (match reference) ----------------
#define BB 2
#define TT 1024
#define DM 1024
#define DD 512
#define NH 8
#define NKV 4
#define HD 64
#define DFF 1280
#define NL 6
#define OUTV 2048

typedef __attribute__((ext_vector_type(16))) __bf16 v16bf;
typedef __attribute__((ext_vector_type(8)))  float  v8f;

// fp32 -> bf16 round-to-nearest-even (explicit, deterministic)
__device__ __forceinline__ __bf16 f2bf(float f) {
  union { float f; unsigned u; } cv; cv.f = f;
  unsigned u = cv.u;
  u += 0x7FFFu + ((u >> 16) & 1u);
  unsigned short hs = (unsigned short)(u >> 16);
  return __builtin_bit_cast(__bf16, hs);
}

__device__ __forceinline__ v8f wmma_bf16(v16bf a, v16bf b, v8f c) {
  // D = A(16x32) * B(32x16) + C, f32 accumulate
  return __builtin_amdgcn_wmma_f32_16x16x32_bf16(
      /*neg_a=*/false, a, /*neg_b=*/false, b,
      /*c_mod=*/(short)0, c, /*reuse_a=*/false, /*reuse_b=*/false);
}

// K index inside a 16-element bf16 fragment (A layout, §7.12.2):
// lanes 0-15 hold K {0..7,16..23}; lanes 16-31 hold K {8..15,24..31}.
__device__ __forceinline__ int frag_k(int e, int lhalf) {
  return e + ((e >= 8) ? 8 : 0) + lhalf * 8;
}

// ================= GEMM: out[M,N] = A[M,K] @ W[K,N] (+ epilogue) =============
// Block tile 64x128, 8 waves as 2x4; each wave computes 32x32 via 4 WMMAs/step.
// EPI: 0 = none, 1 = +emb[tok[row]][col], 2 = +aux[row,col], 3 = silu(aux)*acc
template <int EPI>
__global__ __launch_bounds__(256) void gemm_kernel(
    const float* __restrict__ A, const float* __restrict__ W,
    float* __restrict__ out, int M, int N, int K,
    const float* __restrict__ aux, const int* __restrict__ tok,
    const float* __restrict__ emb) {
  __shared__ __bf16 As[64][40];    // 64 rows x 32 k (+pad, 80B row stride)
  __shared__ __bf16 Wt[128][40];   // transposed: 128 cols x 32 k (+pad)

  const int tid   = threadIdx.x;
  const int wave  = tid >> 5;
  const int lane  = tid & 31;
  const int lhalf = (lane >= 16) ? 1 : 0;
  const int ln    = lane & 15;
  const int wrow  = (wave >> 2) * 32;  // 0 / 32
  const int wcol  = (wave & 3) * 32;   // 0 / 32 / 64 / 96
  const int tm    = blockIdx.y * 64;
  const int tn    = blockIdx.x * 128;

  // staging coords
  const int arow = tid >> 2;          // 0..63
  const int acb  = (tid & 3) * 8;     // 0/8/16/24
  const int wkr  = tid >> 3;          // 0..31
  const int wnb  = (tid & 7) * 16;    // 0..112

  v8f acc00 = {}, acc01 = {}, acc10 = {}, acc11 = {};

  for (int k0 = 0; k0 < K; k0 += 32) {
    // prefetch next tiles into cache while we compute this one
    if (k0 + 32 < K) {
      __builtin_prefetch(A + (size_t)(tm + arow) * K + k0 + 32 + acb, 0, 1);
      __builtin_prefetch(W + (size_t)(k0 + 32 + wkr) * N + tn + wnb, 0, 1);
    }
    __syncthreads();
    {
      // stage A tile (64x32): 8 consecutive k per thread
      const float* ap = A + (size_t)(tm + arow) * K + k0 + acb;
#pragma unroll
      for (int j = 0; j < 8; ++j) As[arow][acb + j] = f2bf(ap[j]);
      // stage W tile (32x128) transposed: 16 consecutive n per thread
      const float* wp = W + (size_t)(k0 + wkr) * N + tn + wnb;
#pragma unroll
      for (int j = 0; j < 16; ++j) Wt[wnb + j][wkr] = f2bf(wp[j]);
    }
    __syncthreads();

    v16bf a0, a1, b0, b1;
#pragma unroll
    for (int e = 0; e < 16; ++e) {
      int kk = frag_k(e, lhalf);
      a0[e] = As[wrow + ln][kk];
      a1[e] = As[wrow + 16 + ln][kk];
      b0[e] = Wt[wcol + ln][kk];
      b1[e] = Wt[wcol + 16 + ln][kk];
    }
    acc00 = wmma_bf16(a0, b0, acc00);
    acc01 = wmma_bf16(a0, b1, acc01);
    acc10 = wmma_bf16(a1, b0, acc10);
    acc11 = wmma_bf16(a1, b1, acc11);
  }

  // epilogue + store (C layout: vgpr r, lane -> row = r + 8*lhalf, col = ln)
#pragma unroll
  for (int r = 0; r < 8; ++r) {
    int row0 = tm + wrow + r + 8 * lhalf;
    int row1 = row0 + 16;
    int c0   = tn + wcol + ln;
    int c1   = c0 + 16;
    float v00 = acc00[r], v01 = acc01[r];
    float v10 = acc10[r], v11 = acc11[r];
    if (EPI == 1) {
      const float* e0 = emb + (size_t)tok[row0] * DD;
      const float* e1 = emb + (size_t)tok[row1] * DD;
      v00 += e0[c0]; v01 += e0[c1];
      v10 += e1[c0]; v11 += e1[c1];
    } else if (EPI == 2) {
      v00 += aux[(size_t)row0 * N + c0];
      v01 += aux[(size_t)row0 * N + c1];
      v10 += aux[(size_t)row1 * N + c0];
      v11 += aux[(size_t)row1 * N + c1];
    } else if (EPI == 3) {
      float g00 = aux[(size_t)row0 * N + c0];
      float g01 = aux[(size_t)row0 * N + c1];
      float g10 = aux[(size_t)row1 * N + c0];
      float g11 = aux[(size_t)row1 * N + c1];
      v00 *= g00 / (1.f + __expf(-g00));
      v01 *= g01 / (1.f + __expf(-g01));
      v10 *= g10 / (1.f + __expf(-g10));
      v11 *= g11 / (1.f + __expf(-g11));
    }
    out[(size_t)row0 * N + c0] = v00;
    out[(size_t)row0 * N + c1] = v01;
    out[(size_t)row1 * N + c0] = v10;
    out[(size_t)row1 * N + c1] = v11;
  }
}

// ======================= RMSNorm (D == 512) ==================================
__global__ __launch_bounds__(256) void rmsnorm_kernel(
    const float* __restrict__ x, const float* __restrict__ w,
    float* __restrict__ y) {
  __shared__ float red[256];
  const int row = blockIdx.x;
  const int tid = threadIdx.x;
  const float* xr = x + (size_t)row * DD;
  float a = xr[tid];
  float b = xr[tid + 256];
  red[tid] = a * a + b * b;
  __syncthreads();
  for (int s = 128; s > 0; s >>= 1) {
    if (tid < s) red[tid] += red[tid + s];
    __syncthreads();
  }
  float rms = rsqrtf(red[0] / (float)DD + 1e-5f);
  y[(size_t)row * DD + tid]       = a * rms * w[tid];
  y[(size_t)row * DD + tid + 256] = b * rms * w[tid + 256];
}

// ======================= RoPE (interleaved pairs) ============================
__global__ __launch_bounds__(256) void rope_kernel(
    float* __restrict__ x, const float* __restrict__ cs,
    const float* __restrict__ sn, int nh) {
  int id = blockIdx.x * blockDim.x + threadIdx.x;
  int total = BB * TT * nh * (HD / 2);
  if (id >= total) return;
  int i = id & 31;                 // HD/2 = 32
  int h = (id >> 5) % nh;
  int t = (id / (32 * nh)) % TT;
  int b = id / (32 * nh * TT);
  float c = cs[t * 32 + i];
  float s = sn[t * 32 + i];
  size_t base = (((size_t)(b * TT + t) * nh + h) * HD) + 2 * i;
  float xe = x[base], xo = x[base + 1];
  x[base]     = xe * c - xo * s;
  x[base + 1] = xe * s + xo * c;
}

// ============ Flash attention: 1 wave per (b, h, 16 query rows) ==============
// q: (B,T,H,HD)  k,v: (B,T,KV,HD)  out: (B,T,H*HD).  GQA n_rep = 2.
__global__ __launch_bounds__(32) void attn_kernel(
    const float* __restrict__ q, const float* __restrict__ k,
    const float* __restrict__ v, float* __restrict__ out) {
  const int qb  = blockIdx.x;
  const int h   = blockIdx.y;
  const int b   = blockIdx.z;
  const int q0  = qb * 16;
  const int kvh = h >> 1;  // H/KV = 2
  const int lane  = threadIdx.x;
  const int lhalf = (lane >= 16) ? 1 : 0;
  const int ln    = lane & 15;

  __shared__ __bf16 Ks[32][HD + 8];
  __shared__ __bf16 Vs[32][HD + 8];
  __shared__ __bf16 Ps[16][40];

  // Q fragments for K-dim halves [0,32) and [32,64)
  v16bf qf0, qf1;
  {
    size_t qbase = (((size_t)(b * TT + q0 + ln)) * NH + h) * HD;
#pragma unroll
    for (int e = 0; e < 16; ++e) {
      int kd = frag_k(e, lhalf);
      qf0[e] = f2bf(q[qbase + kd]);
      qf1[e] = f2bf(q[qbase + kd + 32]);
    }
  }

  v8f o0 = {}, o1 = {}, o2 = {}, o3 = {};
  float m_r[8], l_r[8];
#pragma unroll
  for (int r = 0; r < 8; ++r) { m_r[r] = -1e30f; l_r[r] = 0.f; }

  const int kb_end = (q0 + 15) >> 5;
  for (int kb = 0; kb <= kb_end; ++kb) {
    __syncthreads();
    // stage 32 keys x 64 dims of K and V (bf16)
    for (int idx = lane; idx < 32 * HD; idx += 32) {
      int kr = idx >> 6;
      int kd = idx & 63;
      size_t gk = (((size_t)(b * TT + kb * 32 + kr)) * NKV + kvh) * HD + kd;
      Ks[kr][kd] = f2bf(k[gk]);
      Vs[kr][kd] = f2bf(v[gk]);
    }
    __syncthreads();

    // S = Q @ K^T : two 16x16 accumulators (keys 0..15 and 16..31 of block)
    v8f s0a = {}, s1a = {};
    {
      v16bf bA, bB;
#pragma unroll
      for (int e = 0; e < 16; ++e) {
        int kd = frag_k(e, lhalf);
        bA[e] = Ks[ln][kd];
        bB[e] = Ks[ln][kd + 32];
      }
      s0a = wmma_bf16(qf0, bA, s0a);
      s0a = wmma_bf16(qf1, bB, s0a);
#pragma unroll
      for (int e = 0; e < 16; ++e) {
        int kd = frag_k(e, lhalf);
        bA[e] = Ks[16 + ln][kd];
        bB[e] = Ks[16 + ln][kd + 32];
      }
      s1a = wmma_bf16(qf0, bA, s1a);
      s1a = wmma_bf16(qf1, bB, s1a);
    }

    // online softmax update, row stats reduced across each half-wave
#pragma unroll
    for (int r = 0; r < 8; ++r) {
      int rowq = q0 + r + 8 * lhalf;
      int key0 = kb * 32 + ln;
      int key1 = key0 + 16;
      float s0 = s0a[r] * 0.125f;  // 1/sqrt(64)
      float s1 = s1a[r] * 0.125f;
      if (key0 > rowq) s0 = -1e30f;
      if (key1 > rowq) s1 = -1e30f;
      float mx = fmaxf(s0, s1);
#pragma unroll
      for (int off = 1; off < 16; off <<= 1)
        mx = fmaxf(mx, __shfl_xor(mx, off, 32));
      float mnew  = fmaxf(m_r[r], mx);
      float scale = __expf(m_r[r] - mnew);
      float p0 = __expf(s0 - mnew);
      float p1 = __expf(s1 - mnew);
      float ps = p0 + p1;
#pragma unroll
      for (int off = 1; off < 16; off <<= 1)
        ps += __shfl_xor(ps, off, 32);
      l_r[r] = l_r[r] * scale + ps;
      m_r[r] = mnew;
      o0[r] *= scale; o1[r] *= scale; o2[r] *= scale; o3[r] *= scale;
      Ps[r + 8 * lhalf][ln]      = f2bf(p0);
      Ps[r + 8 * lhalf][ln + 16] = f2bf(p1);
    }
    __syncthreads();

    // O += P(16x32) @ V(32x64) : four 16x16 accumulators over hd groups
    v16bf pf;
#pragma unroll
    for (int e = 0; e < 16; ++e) {
      int kd = frag_k(e, lhalf);
      pf[e] = Ps[ln][kd];
    }
    v16bf vf;
#pragma unroll
    for (int e = 0; e < 16; ++e) vf[e] = Vs[frag_k(e, lhalf)][ln];
    o0 = wmma_bf16(pf, vf, o0);
#pragma unroll
    for (int e = 0; e < 16; ++e) vf[e] = Vs[frag_k(e, lhalf)][16 + ln];
    o1 = wmma_bf16(pf, vf, o1);
#pragma unroll
    for (int e = 0; e < 16; ++e) vf[e] = Vs[frag_k(e, lhalf)][32 + ln];
    o2 = wmma_bf16(pf, vf, o2);
#pragma unroll
    for (int e = 0; e < 16; ++e) vf[e] = Vs[frag_k(e, lhalf)][48 + ln];
    o3 = wmma_bf16(pf, vf, o3);
  }

  // normalize and store: out[(b,t), h*HD + d]
#pragma unroll
  for (int r = 0; r < 8; ++r) {
    int rowq = q0 + r + 8 * lhalf;
    float inv = 1.f / l_r[r];
    size_t base = ((size_t)(b * TT + rowq)) * (NH * HD) + h * HD;
    out[base + ln]      = o0[r] * inv;
    out[base + 16 + ln] = o1[r] * inv;
    out[base + 32 + ln] = o2[r] * inv;
    out[base + 48 + ln] = o3[r] * inv;
  }
}

// ============================== launcher =====================================
extern "C" void kernel_launch(void* const* d_in, const int* in_sizes, int n_in,
                              void* d_out, int out_size, void* d_ws,
                              size_t ws_size, hipStream_t stream) {
  const float* main_hidden = (const float*)d_in[0];
  const int*   prev_token  = (const int*)d_in[1];
  const float* fcos        = (const float*)d_in[2];
  const float* fsin        = (const float*)d_in[3];
  // d_in[4] = mask (causal mask reimplemented in-kernel)
  const float* w_in  = (const float*)d_in[5];
  const float* emb   = (const float*)d_in[6];
  const float* anw   = (const float*)d_in[7];
  const float* wq    = (const float*)d_in[8];
  const float* wk    = (const float*)d_in[9];
  const float* wvw   = (const float*)d_in[10];
  const float* wo    = (const float*)d_in[11];
  const float* fnw   = (const float*)d_in[12];
  const float* wg    = (const float*)d_in[13];
  const float* wu    = (const float*)d_in[14];
  const float* wd    = (const float*)d_in[15];
  const float* finw  = (const float*)d_in[16];
  const float* wh    = (const float*)d_in[17];
  float* outp = (float*)d_out;

  char* ws = (char*)d_ws;
  const size_t MBy = (size_t)1 << 20;
  float* x   = (float*)(ws + 0);          // 4 MB  (B*T*DD)
  float* y   = (float*)(ws + 4 * MBy);    // 4 MB
  float* qb  = (float*)(ws + 8 * MBy);    // 4 MB  (B*T*H*HD)
  float* kb  = (float*)(ws + 12 * MBy);   // 2 MB  (B*T*KV*HD)
  float* vb  = (float*)(ws + 14 * MBy);   // 2 MB
  float* att = (float*)(ws + 16 * MBy);   // 4 MB
  float* gb  = (float*)(ws + 8 * MBy);    // 10 MB (reuses q/k/v after attn)
  float* hb  = (float*)(ws + 18 * MBy);   // 10 MB (reuses att tail; total 28 MB)

  const int M = BB * TT;  // 2048
  dim3 blk(256);

  // x = main_hidden @ w_in + emb[prev_token]
  gemm_kernel<1><<<dim3(DD / 128, M / 64), blk, 0, stream>>>(
      main_hidden, w_in, x, M, DD, DM, nullptr, prev_token, emb);

  for (int l = 0; l < NL; ++l) {
    rmsnorm_kernel<<<M, blk, 0, stream>>>(x, anw + (size_t)l * DD, y);

    gemm_kernel<0><<<dim3((NH * HD) / 128, M / 64), blk, 0, stream>>>(
        y, wq + (size_t)l * DD * NH * HD, qb, M, NH * HD, DD,
        nullptr, nullptr, nullptr);
    gemm_kernel<0><<<dim3((NKV * HD) / 128, M / 64), blk, 0, stream>>>(
        y, wk + (size_t)l * DD * NKV * HD, kb, M, NKV * HD, DD,
        nullptr, nullptr, nullptr);
    gemm_kernel<0><<<dim3((NKV * HD) / 128, M / 64), blk, 0, stream>>>(
        y, wvw + (size_t)l * DD * NKV * HD, vb, M, NKV * HD, DD,
        nullptr, nullptr, nullptr);

    int prq = BB * TT * NH * (HD / 2);
    rope_kernel<<<(prq + 255) / 256, blk, 0, stream>>>(qb, fcos, fsin, NH);
    int prk = BB * TT * NKV * (HD / 2);
    rope_kernel<<<(prk + 255) / 256, blk, 0, stream>>>(kb, fcos, fsin, NKV);

    attn_kernel<<<dim3(TT / 16, NH, BB), dim3(32), 0, stream>>>(qb, kb, vb, att);

    // x = x + att @ wo
    gemm_kernel<2><<<dim3(DD / 128, M / 64), blk, 0, stream>>>(
        att, wo + (size_t)l * NH * HD * DD, x, M, DD, NH * HD,
        x, nullptr, nullptr);

    rmsnorm_kernel<<<M, blk, 0, stream>>>(x, fnw + (size_t)l * DD, y);

    // g = y @ w_gate ; h = silu(g) * (y @ w_up) ; x = x + h @ w_down
    gemm_kernel<0><<<dim3(DFF / 128, M / 64), blk, 0, stream>>>(
        y, wg + (size_t)l * DD * DFF, gb, M, DFF, DD, nullptr, nullptr, nullptr);
    gemm_kernel<3><<<dim3(DFF / 128, M / 64), blk, 0, stream>>>(
        y, wu + (size_t)l * DD * DFF, hb, M, DFF, DD, gb, nullptr, nullptr);
    gemm_kernel<2><<<dim3(DD / 128, M / 64), blk, 0, stream>>>(
        hb, wd + (size_t)l * DFF * DD, x, M, DD, DFF, x, nullptr, nullptr);
  }

  rmsnorm_kernel<<<M, blk, 0, stream>>>(x, finw, y);
  gemm_kernel<0><<<dim3(OUTV / 128, M / 64), blk, 0, stream>>>(
      y, wh, outp, M, OUTV, DD, nullptr, nullptr, nullptr);
}
